// HierNet_19533511262571
// MI455X (gfx1250) — matrix operand order repaired and checked by reading, not compiled, not run on hardware
//
#include <hip/hip_runtime.h>
#include <hip/hip_bf16.h>
#include <math.h>

// ---------------------------------------------------------------------------
// CDNA5 (gfx1250) HierNet. Dense GEMMs on v_wmma_f32_16x16x32_f16 (wave32).
// - Weights pre-converted to f16 once (B loads = strided f16 gathers, no in-loop cvt).
// - Each wave computes a 16x64 tile (4 accumulators, A fragment reused 4x).
// - Branch-free inner loops: A rows clamped, K mult-32 (edge_attr zero-padded),
//   destinations padded to 128-row multiples, A loads 128-bit.
// - Bilinear edge score refactored to node-level GEMM V = xp @ W2,
//   W2[d, s*C+c] = Wb[s,c,d].
// ---------------------------------------------------------------------------

typedef __attribute__((ext_vector_type(16))) _Float16 v16h;
typedef __attribute__((ext_vector_type(8)))  _Float16 v8h;
typedef __attribute__((ext_vector_type(8)))  float    v8f;

#define EPSBN 1e-5f

__device__ __forceinline__ float4 ld4(const float* p) { return *(const float4*)p; }
__device__ __forceinline__ float4 relu4(float4 v)
{
    v.x = fmaxf(v.x, 0.f); v.y = fmaxf(v.y, 0.f);
    v.z = fmaxf(v.z, 0.f); v.w = fmaxf(v.w, 0.f);
    return v;
}
__device__ __forceinline__ void put8(v16h& a, int base, float4 u, float4 v)
{
    a[base + 0] = (_Float16)u.x; a[base + 1] = (_Float16)u.y;
    a[base + 2] = (_Float16)u.z; a[base + 3] = (_Float16)u.w;
    a[base + 4] = (_Float16)v.x; a[base + 5] = (_Float16)v.y;
    a[base + 6] = (_Float16)v.z; a[base + 7] = (_Float16)v.w;
}
// B fragment: column gather, elem i<8 -> k row i, i>=8 -> k row 16+(i-8) (stride ld)
__device__ __forceinline__ v16h ldB(const _Float16* p, int ld)
{
    v16h b;
#pragma unroll
    for (int i = 0; i < 8; ++i) { b[i] = p[(size_t)i * ld]; b[8 + i] = p[(size_t)(16 + i) * ld]; }
    return b;
}

// ---------------------------------------------------------------------------
// Generic WMMA GEMM:  Y[Mpad,Ncol] = act( X[M,K] @ Wh[K,Ncol] + bias )
// Wh in f16. K mult of 32. Ncol mult of 64. Wave = 16 rows x 64 cols.
// grid = (ceil(M/128), Ncol/64), block 256.
// ---------------------------------------------------------------------------
template<int RELU_A, int RELU_OUT, int HAS_BIAS>
__global__ __launch_bounds__(256)
void gemm_wmma(const float* __restrict__ X, const _Float16* __restrict__ Wh,
               const float* __restrict__ bias, float* __restrict__ Y,
               int M, int K, int Ncol)
{
    const int wid  = threadIdx.x >> 5;
    const int lane = threadIdx.x & 31;
    const int lo   = lane & 15;
    const int hi   = lane >> 4;
    const int row0 = blockIdx.x * 128 + wid * 16;
    const int col0 = blockIdx.y * 64;

    const int ra = min(row0 + lo, M - 1);              // clamped A row
    const float* __restrict__ Xr = X + (size_t)ra * K + hi * 8;

    v8f acc[4] = {};
    for (int k0 = 0; k0 < K; k0 += 32) {
        // A fragment: i<8 -> k0+hi*8+i ; i>=8 -> k0+16+hi*8+(i-8)
        float4 a0 = ld4(Xr + k0);
        float4 a1 = ld4(Xr + k0 + 4);
        float4 a2 = ld4(Xr + k0 + 16);
        float4 a3 = ld4(Xr + k0 + 20);
        if (RELU_A) { a0 = relu4(a0); a1 = relu4(a1); a2 = relu4(a2); a3 = relu4(a3); }
        v16h a;
        put8(a, 0, a0, a1);
        put8(a, 8, a2, a3);
        const _Float16* __restrict__ Wk = Wh + (size_t)(k0 + hi * 8) * Ncol + col0 + lo;
#pragma unroll
        for (int t = 0; t < 4; ++t) {
            const v16h b = ldB(Wk + 16 * t, Ncol);
            acc[t] = __builtin_amdgcn_wmma_f32_16x16x32_f16(false, a, false, b,
                                                            (short)0, acc[t], false, false);
        }
    }
    // D: elem i -> row = row0 + hi*8 + i, col = col0 + 16*t + lo (padded store)
#pragma unroll
    for (int t = 0; t < 4; ++t) {
        const float bv = HAS_BIAS ? bias[col0 + 16 * t + lo] : 0.f;
        float* __restrict__ Yr = Y + (size_t)(row0 + hi * 8) * Ncol + col0 + 16 * t + lo;
#pragma unroll
        for (int i = 0; i < 8; ++i) {
            float o = acc[t][i] + bv;
            if (RELU_OUT) o = fmaxf(o, 0.f);
            Yr[(size_t)i * Ncol] = o;
        }
    }
}

// ---------------------------------------------------------------------------
// Gate GEMM: beta = relu( [h, relu(agg), h-relu(agg)] @ gate_w + gate_b )
// K = 384 as 3 segments of 128 (runtime seg loop keeps pressure bounded).
// Ncol = 128, wave = 16x64, grid = (ceil(M/128), 2).
// ---------------------------------------------------------------------------
__global__ __launch_bounds__(256)
void gemm_gate_wmma(const float* __restrict__ h, const float* __restrict__ agg,
                    const _Float16* __restrict__ Wg, const float* __restrict__ bias,
                    float* __restrict__ beta, int M)
{
    const int wid  = threadIdx.x >> 5;
    const int lane = threadIdx.x & 31;
    const int lo   = lane & 15;
    const int hi   = lane >> 4;
    const int row0 = blockIdx.x * 128 + wid * 16;
    const int col0 = blockIdx.y * 64;

    const int ra = min(row0 + lo, M - 1);
    const float* __restrict__ Hr = h   + (size_t)ra * 128 + hi * 8;
    const float* __restrict__ Gr = agg + (size_t)ra * 128 + hi * 8;

    v8f acc[4] = {};
    for (int seg = 0; seg < 3; ++seg) {
        for (int ko = 0; ko < 128; ko += 32) {
            float4 a0, a1, a2, a3;
            if (seg == 0) {
                a0 = ld4(Hr + ko);      a1 = ld4(Hr + ko + 4);
                a2 = ld4(Hr + ko + 16); a3 = ld4(Hr + ko + 20);
            } else if (seg == 1) {
                a0 = relu4(ld4(Gr + ko));      a1 = relu4(ld4(Gr + ko + 4));
                a2 = relu4(ld4(Gr + ko + 16)); a3 = relu4(ld4(Gr + ko + 20));
            } else {
                const float4 h0 = ld4(Hr + ko),      h1 = ld4(Hr + ko + 4);
                const float4 h2 = ld4(Hr + ko + 16), h3 = ld4(Hr + ko + 20);
                const float4 g0 = relu4(ld4(Gr + ko)),      g1 = relu4(ld4(Gr + ko + 4));
                const float4 g2 = relu4(ld4(Gr + ko + 16)), g3 = relu4(ld4(Gr + ko + 20));
                a0 = make_float4(h0.x - g0.x, h0.y - g0.y, h0.z - g0.z, h0.w - g0.w);
                a1 = make_float4(h1.x - g1.x, h1.y - g1.y, h1.z - g1.z, h1.w - g1.w);
                a2 = make_float4(h2.x - g2.x, h2.y - g2.y, h2.z - g2.z, h2.w - g2.w);
                a3 = make_float4(h3.x - g3.x, h3.y - g3.y, h3.z - g3.z, h3.w - g3.w);
            }
            v16h a;
            put8(a, 0, a0, a1);
            put8(a, 8, a2, a3);
            const _Float16* __restrict__ Wk =
                Wg + (size_t)(seg * 128 + ko + hi * 8) * 128 + col0 + lo;
#pragma unroll
            for (int t = 0; t < 4; ++t) {
                const v16h b = ldB(Wk + 16 * t, 128);
                acc[t] = __builtin_amdgcn_wmma_f32_16x16x32_f16(false, a, false, b,
                                                                (short)0, acc[t], false, false);
            }
        }
    }
#pragma unroll
    for (int t = 0; t < 4; ++t) {
        const float bv = bias[col0 + 16 * t + lo];
        float* __restrict__ Br = beta + (size_t)(row0 + hi * 8) * 128 + col0 + 16 * t + lo;
#pragma unroll
        for (int i = 0; i < 8; ++i)
            Br[(size_t)i * 128] = fmaxf(acc[t][i] + bv, 0.f);
    }
}

// ---------------------------------------------------------------------------
// Column-wise BN statistics over [M,128]. One block per column.
// ---------------------------------------------------------------------------
__global__ void colstats(const float* __restrict__ Y, int M,
                         float* __restrict__ mean, float* __restrict__ var)
{
    const int c = blockIdx.x;
    __shared__ float ss[256], sq[256];
    float s = 0.f, q = 0.f;
    for (int r = threadIdx.x; r < M; r += blockDim.x) {
        const float v = Y[(size_t)r * 128 + c];
        s += v; q += v * v;
    }
    ss[threadIdx.x] = s; sq[threadIdx.x] = q;
    __syncthreads();
    for (int st = 128; st > 0; st >>= 1) {
        if ((int)threadIdx.x < st) {
            ss[threadIdx.x] += ss[threadIdx.x + st];
            sq[threadIdx.x] += sq[threadIdx.x + st];
        }
        __syncthreads();
    }
    if (threadIdx.x == 0) {
        const float mu = ss[0] / (float)M;
        mean[c] = mu;
        var[c]  = sq[0] / (float)M - mu * mu;
    }
}

__global__ void bn_apply_relu(float* __restrict__ y, const float* __restrict__ mean,
                              const float* __restrict__ var, const float* __restrict__ g,
                              const float* __restrict__ b, size_t n)
{
    const size_t i = (size_t)blockIdx.x * blockDim.x + threadIdx.x;
    if (i >= n) return;
    const int c = (int)(i & 127);
    const float v = y[i];
    y[i] = fmaxf((v - mean[c]) * rsqrtf(var[c] + EPSBN) * g[c] + b[c], 0.f);
}

__global__ void fill_kernel(float* __restrict__ p, size_t n, float v)
{
    const size_t i = (size_t)blockIdx.x * blockDim.x + threadIdx.x;
    if (i < n) p[i] = v;
}

__global__ void relu_ip(float* __restrict__ p, size_t n)
{
    const size_t i = (size_t)blockIdx.x * blockDim.x + threadIdx.x;
    if (i < n) p[i] = fmaxf(p[i], 0.f);
}

// f32 -> f16 weight conversion
__global__ void cvt16(const float* __restrict__ s, _Float16* __restrict__ d, size_t n)
{
    const size_t i = (size_t)blockIdx.x * blockDim.x + threadIdx.x;
    if (i < n) d[i] = (_Float16)s[i];
}

// zero-pad K dimension: dst[M,32] <- src[M,16] (cols 16..31 = 0)
__global__ void pad_k16(const float* __restrict__ src, float* __restrict__ dst, int M)
{
    const int t = blockIdx.x * blockDim.x + threadIdx.x;
    if (t >= M * 32) return;
    const int r = t >> 5, c = t & 31;
    dst[t] = (c < 16) ? src[r * 16 + c] : 0.f;
}

// zero-pad + convert weight rows: dst[32,128] f16 <- src[16,128] f32
__global__ void pad_w16h(const float* __restrict__ src, _Float16* __restrict__ dst)
{
    const int t = blockIdx.x * blockDim.x + threadIdx.x;
    if (t >= 32 * 128) return;
    dst[t] = (t < 16 * 128) ? (_Float16)src[t] : (_Float16)0.f;
}

// W2h[l][d][s*128+c] = (f16) wb[l][s][c][d]   (for V = xp @ W2)
__global__ void build_w2h(const float* __restrict__ wb, _Float16* __restrict__ W2, int L)
{
    const int t = blockIdx.x * blockDim.x + threadIdx.x;
    const int total = L << 17;                          // L*8*128*128
    if (t >= total) return;
    const int d = t & 127, c = (t >> 7) & 127, s = (t >> 14) & 7, l = t >> 17;
    W2[((size_t)l << 17) + (size_t)d * 1024 + s * 128 + c] =
        (_Float16)wb[((size_t)((l * 8 + s) * 128 + c)) * 128 + d];
}

// ---------------------------------------------------------------------------
// Per-edge finalize: one wave32 per edge. Lane covers c in {lane,+32,+64,+96}.
// score[s] = xi . V[src, s*128:] ; block[s] = [xi,ep,xj] . Wl[:,s]
// alpha = tanh(score+block+wlb); msg = max(xj,ep)*alpha[c>>4]; atomic into dst.
// ---------------------------------------------------------------------------
__global__ void edge_msg(const float* __restrict__ xp, const float* __restrict__ ep,
                         const float* __restrict__ V,  const float* __restrict__ Wl,
                         const float* __restrict__ wlb, const int* __restrict__ src,
                         const int* __restrict__ dst, float* __restrict__ agg, int Ecnt)
{
    const int wid  = threadIdx.x >> 5;
    const int lane = threadIdx.x & 31;
    const int e = blockIdx.x * 8 + wid;
    if (e >= Ecnt) return;
    const int sI = src[e], dI = dst[e];
    const float* xi = xp + (size_t)dI * 128;
    const float* xj = xp + (size_t)sI * 128;
    const float* ev = ep + (size_t)e  * 128;
    const float* Ve = V  + (size_t)sI * 1024;

    float xiv[4], xjv[4], epv[4];
#pragma unroll
    for (int t = 0; t < 4; ++t) {
        const int c = lane + t * 32;
        xiv[t] = xi[c]; xjv[t] = xj[c]; epv[t] = ev[c];
    }
    float alpha[8];
#pragma unroll
    for (int s = 0; s < 8; ++s) {
        float p = 0.f;
#pragma unroll
        for (int t = 0; t < 4; ++t) {
            const int c = lane + t * 32;
            p += xiv[t] * Ve[s * 128 + c];         // bilinear score
            p += xiv[t] * Wl[(c)       * 8 + s];   // block: xi part
            p += epv[t] * Wl[(128 + c) * 8 + s];   // block: ep part
            p += xjv[t] * Wl[(256 + c) * 8 + s];   // block: xj part
        }
        for (int m = 16; m >= 1; m >>= 1) p += __shfl_xor(p, m, 32);
        alpha[s] = tanhf(p + wlb[s]);
    }
#pragma unroll
    for (int t = 0; t < 4; ++t) {
        const int c = lane + t * 32;
        const float m = fmaxf(xjv[t], epv[t]) * alpha[c >> 4];
        atomicAdd(&agg[(size_t)dI * 128 + c], m);
    }
}

__global__ void gate_combine(float* __restrict__ h, const float* __restrict__ beta,
                             const float* __restrict__ agg, size_t n)
{
    const size_t i = (size_t)blockIdx.x * blockDim.x + threadIdx.x;
    if (i >= n) return;
    const float b  = beta[i];
    const float hh = fmaxf(agg[i], 0.f);
    h[i] = b * h[i] + (1.f - b) * hh;
}

__global__ void pool_add(const float* __restrict__ fx, const int* __restrict__ cl,
                         float* __restrict__ pool, int NF)
{
    const int t = blockIdx.x * blockDim.x + threadIdx.x;
    if (t >= NF * 128) return;
    const int i = t >> 7, c = t & 127;
    atomicAdd(&pool[(size_t)cl[i] * 128 + c], fx[t]);
}

__global__ void build_perm(const int* __restrict__ cl, int* __restrict__ perm,
                           int NF, const int* __restrict__ ncl)
{
    const int f = blockIdx.x * blockDim.x + threadIdx.x;
    if (f >= NF || f >= ncl[0]) return;
    int lo = 0, hi = NF;
    while (lo < hi) { const int mid = (lo + hi) >> 1; if (cl[mid] < f) lo = mid + 1; else hi = mid; }
    perm[f] = lo;
}

// a[r,h] = sum_c Hm[r, h*128+c] * A[h,c]
__global__ void att_row(const float* __restrict__ Hm, const float* __restrict__ A,
                        float* __restrict__ out, int rows)
{
    const int t = blockIdx.x * blockDim.x + threadIdx.x;
    if (t >= rows * 8) return;
    const int r = t >> 3, h = t & 7;
    float s = 0.f;
    for (int c = 0; c < 128; c += 4) {
        const float4 a = ld4(Hm + (size_t)r * 1024 + h * 128 + c);
        const float4 b = ld4(A + h * 128 + c);
        s += a.x * b.x + a.y * b.y + a.z * b.z + a.w * b.w;
    }
    out[t] = s;
}

__device__ inline void atomicMaxF(float* addr, float val)
{
    int* ia = (int*)addr;
    int cur = __float_as_int(*addr);
    while (__int_as_float(cur) < val) {
        const int prev = atomicCAS(ia, cur, __float_as_int(val));
        if (prev == cur) break;
        cur = prev;
    }
}

__global__ void gat_el_max(const float* __restrict__ a_src, const float* __restrict__ a_dst,
                           const int* __restrict__ perm, float* __restrict__ el,
                           float* __restrict__ mvals, int NF, const int* __restrict__ ncl)
{
    const int t = blockIdx.x * blockDim.x + threadIdx.x;
    if (t >= NF * 8) return;
    const int f = t >> 3, h = t & 7;
    if (f >= ncl[0]) return;
    float v = a_src[t] + a_dst[perm[f] * 8 + h];
    v = (v > 0.f) ? v : 0.01f * v;               // leaky_relu(0.01)
    el[t] = v;
    atomicMaxF(&mvals[perm[f] * 8 + h], v);
}

__global__ void gat_ex(float* __restrict__ el, const float* __restrict__ mvals,
                       float* __restrict__ sden, const int* __restrict__ perm,
                       int NF, const int* __restrict__ ncl)
{
    const int t = blockIdx.x * blockDim.x + threadIdx.x;
    if (t >= NF * 8) return;
    const int f = t >> 3, h = t & 7;
    if (f >= ncl[0]) return;
    const float v = expf(el[t] - mvals[perm[f] * 8 + h]);
    el[t] = v;
    atomicAdd(&sden[perm[f] * 8 + h], v);
}

__global__ void gat_agg(const float* __restrict__ el, const float* __restrict__ sden,
                        const int* __restrict__ perm, const float* __restrict__ hsrc,
                        float* __restrict__ agg2, int NF, const int* __restrict__ ncl)
{
    const int t = blockIdx.x * blockDim.x + threadIdx.x;
    if (t >= NF * 1024) return;
    const int f = t >> 10;
    if (f >= ncl[0]) return;
    const int rem = t & 1023, h = rem >> 7;
    const float att = el[f * 8 + h] / sden[perm[f] * 8 + h];
    atomicAdd(&agg2[(size_t)perm[f] * 1024 + rem], hsrc[t] * att);
}

__global__ void final_out(const float* __restrict__ h_atom, const float* __restrict__ agg2,
                          const float* __restrict__ gat_bias, const float* __restrict__ out_w,
                          const float* __restrict__ out_b, float* __restrict__ out, int N)
{
    const int n = blockIdx.x * blockDim.x + threadIdx.x;
    if (n >= N) return;
    float o = out_b[0];
    for (int c = 0; c < 128; ++c) {
        const float mol = fmaxf(h_atom[(size_t)n * 128 + c], 0.f);
        float s = 0.f;
        for (int h2 = 0; h2 < 8; ++h2) s += agg2[(size_t)n * 1024 + h2 * 128 + c];
        const float fm = fmaxf(s * 0.125f + gat_bias[c], 0.f);
        o += mol * out_w[c] + fm * out_w[128 + c];
    }
    out[n] = o;
}

// ---------------------------------------------------------------------------
extern "C" void kernel_launch(void* const* d_in, const int* in_sizes, int n_in,
                              void* d_out, int out_size, void* d_ws, size_t ws_size,
                              hipStream_t stream)
{
    (void)n_in; (void)out_size; (void)ws_size;
    const float* x        = (const float*)d_in[0];
    const float* eattr    = (const float*)d_in[1];
    const float* fx_in    = (const float*)d_in[2];
    const float* feattr   = (const float*)d_in[3];
    const float* la_w     = (const float*)d_in[4];
    const float* la_b     = (const float*)d_in[5];
    const float* lb_w     = (const float*)d_in[6];
    const float* lb_b     = (const float*)d_in[7];
    const float* bn1_g    = (const float*)d_in[8];
    const float* bn1_b    = (const float*)d_in[9];
    const float* bn2_g    = (const float*)d_in[10];
    const float* bn2_b    = (const float*)d_in[11];
    const float* wn       = (const float*)d_in[12];
    const float* wb       = (const float*)d_in[13];
    const float* wl       = (const float*)d_in[14];
    const float* wlb      = (const float*)d_in[15];
    const float* gate_w   = (const float*)d_in[16];
    const float* gate_b   = (const float*)d_in[17];
    const float* gat_wsrc = (const float*)d_in[18];
    const float* gat_wdst = (const float*)d_in[19];
    const float* gat_asrc = (const float*)d_in[20];
    const float* gat_adst = (const float*)d_in[21];
    const float* gat_bias = (const float*)d_in[22];
    const float* out_w    = (const float*)d_in[23];
    const float* out_b    = (const float*)d_in[24];
    const int*   ei       = (const int*)d_in[25];
    const int*   fei      = (const int*)d_in[26];
    const int*   cluster  = (const int*)d_in[27];
    const int*   ncl      = (const int*)d_in[28];

    const int C   = in_sizes[5];             // 128
    const int CIN = in_sizes[4] / C;         // 128
    const int N   = in_sizes[0] / CIN;
    const int NF  = in_sizes[2] / CIN;
    const int E   = in_sizes[25] / 2;
    const int EF  = in_sizes[26] / 2;
    const int L   = in_sizes[12] / (C * C);  // 3

    auto rup = [](long v) { return (size_t)((v + 127) & ~127L); };
    const size_t Np = rup(N), Ep = rup(E), NFp = rup(NF), EFp = rup(EF);

    // workspace carve-out (float units); GEMM dests padded to 128-row mult.
    float* ws = (float*)d_ws;
    size_t off = 0;
    auto take = [&](size_t n) { float* p = ws + off; off += n; return p; };
    auto takeh = [&](size_t n) { return (_Float16*)take((n + 1) / 2); };
    float* h_atom    = take(Np * 128);
    float* ea_atom   = take(Ep * 128);
    float* h_frag    = take(NFp * 128);
    float* ea_frag   = take(EFp * 128);
    float* xp        = take(Np * 128);
    float* ep        = take(Ep * 128);
    float* big       = take(Np * 1024);        // V -> h_dst -> agg2 (time-shared)
    float* agg       = take(Np * 128);
    float* beta      = take(Np * 128);
    float* mean      = take(128);
    float* var       = take(128);
    float* frag_pool = take(NFp * 128);
    float* hsrc      = take(NFp * 1024);
    float* a_src     = take(NFp * 8);
    float* a_dst     = take(Np * 8);
    float* elbuf     = take(NFp * 8);
    float* mvals     = take(Np * 8);
    float* sden      = take(Np * 8);
    float* ea_pad    = take(Ep * 32);          // K-padded edge_attr (f32 A operand)
    float* fea_pad   = take(EFp * 32);
    // f16 weights
    _Float16* laH    = takeh((size_t)CIN * 128);
    _Float16* lbH    = takeh(32 * 128);
    _Float16* wnH    = takeh((size_t)L * 128 * 128);
    _Float16* gateH  = takeh(384 * 128);
    _Float16* gsrcH  = takeh(128 * 1024);
    _Float16* gdstH  = takeh(128 * 1024);
    _Float16* W2H    = takeh((size_t)L * 128 * 1024);
    int*      perm   = (int*)take(NFp);

    const dim3 blk(256);
    auto cdiv = [](long a, long b) { return (unsigned)((a + b - 1) / b); };

    // ---- weight prep: f16 conversions, Wb reshape, K pads -----------------
    cvt16<<<cdiv((long)CIN * 128, 256), blk, 0, stream>>>(la_w, laH, (size_t)CIN * 128);
    pad_w16h<<<cdiv(32 * 128, 256), blk, 0, stream>>>(lb_w, lbH);
    cvt16<<<cdiv((long)L * 128 * 128, 256), blk, 0, stream>>>(wn, wnH, (size_t)L * 128 * 128);
    cvt16<<<cdiv(384 * 128, 256), blk, 0, stream>>>(gate_w, gateH, 384 * 128);
    cvt16<<<cdiv(128 * 1024, 256), blk, 0, stream>>>(gat_wsrc, gsrcH, 128 * 1024);
    cvt16<<<cdiv(128 * 1024, 256), blk, 0, stream>>>(gat_wdst, gdstH, 128 * 1024);
    build_w2h<<<cdiv((long)L << 17, 256), blk, 0, stream>>>(wb, W2H, L);
    pad_k16<<<cdiv((long)E * 32, 256), blk, 0, stream>>>(eattr, ea_pad, E);
    pad_k16<<<cdiv((long)EF * 32, 256), blk, 0, stream>>>(feattr, fea_pad, EF);

    // ---- input projections ------------------------------------------------
    gemm_wmma<0,0,1><<<dim3(cdiv(N, 128), 2), blk, 0, stream>>>(x, laH, la_b, h_atom, N, CIN, 128);
    colstats<<<128, 256, 0, stream>>>(h_atom, N, mean, var);
    bn_apply_relu<<<cdiv((long)N * 128, 256), blk, 0, stream>>>(h_atom, mean, var, bn1_g, bn1_b, (size_t)N * 128);

    gemm_wmma<0,0,1><<<dim3(cdiv(E, 128), 2), blk, 0, stream>>>(ea_pad, lbH, lb_b, ea_atom, E, 32, 128);
    colstats<<<128, 256, 0, stream>>>(ea_atom, E, mean, var);
    bn_apply_relu<<<cdiv((long)E * 128, 256), blk, 0, stream>>>(ea_atom, mean, var, bn2_g, bn2_b, (size_t)E * 128);

    gemm_wmma<0,1,1><<<dim3(cdiv(NF, 128), 2), blk, 0, stream>>>(fx_in, laH, la_b, h_frag, NF, CIN, 128);

    gemm_wmma<0,0,1><<<dim3(cdiv(EF, 128), 2), blk, 0, stream>>>(fea_pad, lbH, lb_b, ea_frag, EF, 32, 128);
    colstats<<<128, 256, 0, stream>>>(ea_frag, EF, mean, var);
    bn_apply_relu<<<cdiv((long)EF * 128, 256), blk, 0, stream>>>(ea_frag, mean, var, bn2_g, bn2_b, (size_t)EF * 128);

    // ---- NTN stacks (atom graph, fragment graph) --------------------------
    struct G { float* h; float* ea; int M; int Ecnt; const int* src; const int* dst; };
    G gs[2] = { { h_atom, ea_atom, N,  E,  ei,  ei  + E  },
                { h_frag, ea_frag, NF, EF, fei, fei + EF } };
    for (int g = 0; g < 2; ++g) {
        const G& gg = gs[g];
        for (int l = 0; l < L; ++l) {
            const _Float16* WnH = wnH + (size_t)l * 128 * 128;
            const float*    Wl  = wl  + (size_t)l * 384 * 8;
            const float*    blb = wlb + (size_t)l * 8;
            gemm_wmma<0,0,0><<<dim3(cdiv(gg.M, 128), 2), blk, 0, stream>>>(gg.h, WnH, nullptr, xp, gg.M, 128, 128);
            gemm_wmma<0,0,0><<<dim3(cdiv(gg.Ecnt, 128), 2), blk, 0, stream>>>(gg.ea, WnH, nullptr, ep, gg.Ecnt, 128, 128);
            gemm_wmma<0,0,0><<<dim3(cdiv(gg.M, 128), 16), blk, 0, stream>>>(xp, W2H + ((size_t)l << 17), nullptr, big, gg.M, 128, 1024);
            fill_kernel<<<cdiv((long)gg.M * 128, 256), blk, 0, stream>>>(agg, (size_t)gg.M * 128, 0.f);
            edge_msg<<<cdiv(gg.Ecnt, 8), blk, 0, stream>>>(xp, ep, big, Wl, blb, gg.src, gg.dst, agg, gg.Ecnt);
            gemm_gate_wmma<<<dim3(cdiv(gg.M, 128), 2), blk, 0, stream>>>(gg.h, agg, gateH, gate_b, beta, gg.M);
            gate_combine<<<cdiv((long)gg.M * 128, 256), blk, 0, stream>>>(gg.h, beta, agg, (size_t)gg.M * 128);
        }
    }

    // ---- fragment pooling + perm ------------------------------------------
    fill_kernel<<<cdiv((long)NF * 128, 256), blk, 0, stream>>>(frag_pool, (size_t)NF * 128, 0.f);
    pool_add<<<cdiv((long)NF * 128, 256), blk, 0, stream>>>(h_frag, cluster, frag_pool, NF);
    relu_ip<<<cdiv((long)NF * 128, 256), blk, 0, stream>>>(frag_pool, (size_t)NF * 128);
    build_perm<<<cdiv(NF, 256), blk, 0, stream>>>(cluster, perm, NF, ncl);

    // ---- GAT cross attention ----------------------------------------------
    gemm_wmma<0,0,0><<<dim3(cdiv(NF, 128), 16), blk, 0, stream>>>(frag_pool, gsrcH, nullptr, hsrc, NF, 128, 1024);
    att_row<<<cdiv((long)NF * 8, 256), blk, 0, stream>>>(hsrc, gat_asrc, a_src, NF);

    gemm_wmma<1,0,0><<<dim3(cdiv(N, 128), 16), blk, 0, stream>>>(h_atom, gdstH, nullptr, big, N, 128, 1024); // relu_a: mol_vec
    att_row<<<cdiv((long)N * 8, 256), blk, 0, stream>>>(big, gat_adst, a_dst, N);

    fill_kernel<<<cdiv((long)N * 8, 256), blk, 0, stream>>>(mvals, (size_t)N * 8, -3e38f);
    fill_kernel<<<cdiv((long)N * 8, 256), blk, 0, stream>>>(sden, (size_t)N * 8, 0.f);
    gat_el_max<<<cdiv((long)NF * 8, 256), blk, 0, stream>>>(a_src, a_dst, perm, elbuf, mvals, NF, ncl);
    gat_ex<<<cdiv((long)NF * 8, 256), blk, 0, stream>>>(elbuf, mvals, sden, perm, NF, ncl);

    fill_kernel<<<cdiv((long)N * 1024, 256), blk, 0, stream>>>(big, (size_t)N * 1024, 0.f); // agg2
    gat_agg<<<cdiv((long)NF * 1024, 256), blk, 0, stream>>>(elbuf, sden, perm, hsrc, big, NF, ncl);

    final_out<<<cdiv(N, 256), blk, 0, stream>>>(h_atom, big, gat_bias, out_w, out_b, (float*)d_out, N);
}
